// StyledConv1d_41351945125947
// MI455X (gfx1250) — matrix operand order repaired and checked by reading, not compiled, not run on hardware
//
#include <hip/hip_runtime.h>

typedef __attribute__((ext_vector_type(16))) _Float16 v16h;
typedef __attribute__((ext_vector_type(8)))  _Float16 v8h;
typedef __attribute__((ext_vector_type(8)))  float    v8f;

#define B_      16
#define CIN_    256
#define COUT_   256
#define L_      8192
#define SDIM_   512
#define K_      3
#define LIN_SCALE_  0.044194173824159216f   /* 1/sqrt(512) */
#define CONV_SCALE_ 0.020833333333333332f   /* 1/sqrt(256*9) = 1/48 */

// ---------------------------------------------------------------------------
// Kernel 1: s[b][ci] = style[b] . modW[ci] * LIN_SCALE + bias[ci]
// grid: (B_), block: 256 (one thread per ci)
// ---------------------------------------------------------------------------
__global__ __launch_bounds__(256)
void style_mod_kernel(const float* __restrict__ style,
                      const float* __restrict__ mw,
                      const float* __restrict__ mb,
                      float* __restrict__ s)
{
    const int b  = blockIdx.x;
    const int ci = threadIdx.x;
    const float* st = style + (size_t)b * SDIM_;
    const float* w  = mw + (size_t)ci * SDIM_;
    float acc = 0.0f;
    #pragma unroll 4
    for (int d = 0; d < SDIM_; d += 4) {
        float4 a  = *(const float4*)(st + d);
        float4 ww = *(const float4*)(w + d);
        acc += a.x * ww.x + a.y * ww.y + a.z * ww.z + a.w * ww.w;
    }
    s[b * CIN_ + ci] = acc * LIN_SCALE_ + mb[ci];
}

// ---------------------------------------------------------------------------
// Kernel 2: per-(b,co) modulate + demodulate, emit f16 weights wf[b][co][k][ci]
// One wave (32 lanes) per (b,co). grid: 512 blocks x 256 threads (8 waves/blk)
// ---------------------------------------------------------------------------
__global__ __launch_bounds__(256)
void demod_kernel(const float* __restrict__ weight,   // [COUT][CIN][K] (k fastest)
                  const float* __restrict__ s,        // [B][CIN]
                  _Float16* __restrict__ wf)          // [B][COUT][K][CIN] halves
{
    const int wid  = blockIdx.x * 8 + (threadIdx.x >> 5);
    const int lane = threadIdx.x & 31;
    const int b  = wid >> 8;
    const int co = wid & 255;

    const float* wrow = weight + (size_t)co * (CIN_ * K_);
    const float* sb   = s + b * CIN_;

    float m[24];
    float ssum = 0.0f;
    #pragma unroll
    for (int j = 0; j < 24; ++j) {
        const int e  = lane + 32 * j;          // 0..767
        const int ci = e / 3;
        const float v = CONV_SCALE_ * wrow[e] * sb[ci];
        m[j] = v;
        ssum += v * v;
    }
    // wave32 butterfly reduction
    for (int off = 16; off > 0; off >>= 1)
        ssum += __shfl_xor(ssum, off, 32);
    const float dmod = rsqrtf(ssum + 1e-8f);

    _Float16* out = wf + ((size_t)(b * COUT_ + co)) * (K_ * CIN_);
    #pragma unroll
    for (int j = 0; j < 24; ++j) {
        const int e  = lane + 32 * j;
        const int ci = e / 3;
        const int k  = e - 3 * ci;
        out[k * CIN_ + ci] = (_Float16)(m[j] * dmod);
    }
}

// ---------------------------------------------------------------------------
// Kernel 3: implicit-GEMM conv via v_wmma_f32_16x16x32_f16.
// grid: (L/64, B). block: 256 threads = 8 waves.
// Workgroup computes out[b][0..255][l0..l0+64). Wave w owns couts [32w,32w+32).
// LDS holds x[b][:, l0-1 .. l0+64] transposed to [l][ci] as f16.
// ---------------------------------------------------------------------------
#define LBLK  64
#define ROWH  264                 /* halves per LDS row: 256 + 8 pad (16B) */

__global__ __launch_bounds__(256)
void conv_wmma_kernel(const float* __restrict__ x,      // [B][CIN][L]
                      const _Float16* __restrict__ wf,  // [B][COUT][K][CIN]
                      float* __restrict__ out)          // [B][COUT][L]
{
    __shared__ _Float16 xt[(LBLK + 2) * ROWH];          // 34,848 B

    const int b   = blockIdx.y;
    const int l0  = blockIdx.x * LBLK;
    const int tid = threadIdx.x;

    // ---- stage x tile, fp32 -> f16, transposed to [l][ci] with halo ----
    // Only rows 0 (l0-1) and LBLK+1 (l0+LBLK) can be out of range; handle
    // them branch-free with clamped addresses. Interior rows are always
    // valid -> unconditional float4 loads (no per-element waits).
    {
        const int ci = tid;                             // 256 threads = 256 ci rows
        const float* xp = x + ((size_t)(b * CIN_ + ci)) * L_;

        const bool lo_ok = (l0 > 0);
        const bool hi_ok = (l0 + LBLK < L_);
        float vlo = xp[lo_ok ? (l0 - 1) : 0];
        float vhi = xp[hi_ok ? (l0 + LBLK) : 0];
        xt[0 * ROWH + ci]          = (_Float16)(lo_ok ? vlo : 0.0f);
        xt[(LBLK + 1) * ROWH + ci] = (_Float16)(hi_ok ? vhi : 0.0f);

        #pragma unroll 4
        for (int j = 0; j < LBLK; j += 4) {
            float4 v = *(const float4*)(xp + l0 + j);   // l0+j .. l0+j+3, always valid
            xt[(j + 1) * ROWH + ci] = (_Float16)v.x;
            xt[(j + 2) * ROWH + ci] = (_Float16)v.y;
            xt[(j + 3) * ROWH + ci] = (_Float16)v.z;
            xt[(j + 4) * ROWH + ci] = (_Float16)v.w;
        }
    }
    __syncthreads();

    const int wave = tid >> 5;
    const int lane = tid & 31;
    const int lr   = lane & 15;     // row/col within 16
    const int hi   = lane >> 4;     // half-wave select
    const int co_base = wave * 32;

    v8f acc[2][4] = {};

    union H16 { v16h v; v8h h[2]; };

    for (int ci0 = 0; ci0 < CIN_; ci0 += 32) {
        #pragma unroll
        for (int k = 0; k < K_; ++k) {
            // A fragments: 16x32 f16, M = co, K = ci chunk
            v16h afrag[2];
            #pragma unroll
            for (int ct = 0; ct < 2; ++ct) {
                const _Float16* ap = wf
                    + ((size_t)(b * COUT_ + co_base + ct * 16 + lr)) * (K_ * CIN_)
                    + k * CIN_ + ci0 + hi * 8;
                H16 t;
                t.h[0] = *(const v8h*)(ap);        // K local: hi*8 + 0..7
                t.h[1] = *(const v8h*)(ap + 16);   // K local: hi*8 + 16..23
                afrag[ct] = t.v;
            }
            // B fragments: 32x16 f16, K = ci chunk, N = l within subtile
            v16h bfrag[4];
            #pragma unroll
            for (int lt = 0; lt < 4; ++lt) {
                const int row = lt * 16 + lr + k;  // halo-shifted l index
                const _Float16* bp = xt + row * ROWH + ci0 + hi * 8;
                H16 t;
                t.h[0] = *(const v8h*)(bp);        // K local: hi*8 + 0..7
                t.h[1] = *(const v8h*)(bp + 16);   // K local: hi*8 + 16..23
                bfrag[lt] = t.v;
            }
            #pragma unroll
            for (int ct = 0; ct < 2; ++ct)
                #pragma unroll
                for (int lt = 0; lt < 4; ++lt)
                    acc[ct][lt] = __builtin_amdgcn_wmma_f32_16x16x32_f16(
                        /*neg_a=*/false, afrag[ct],
                        /*neg_b=*/false, bfrag[lt],
                        /*c_mod=*/(short)0, acc[ct][lt],
                        /*reuse_a=*/false, /*reuse_b=*/false);
        }
    }

    // ---- writeback: D layout lane<16 -> M=r, lane>=16 -> M=8+r ----
    #pragma unroll
    for (int ct = 0; ct < 2; ++ct) {
        #pragma unroll
        for (int lt = 0; lt < 4; ++lt) {
            float* op = out
                + ((size_t)(b * COUT_ + co_base + ct * 16 + hi * 8)) * L_
                + l0 + lt * 16 + lr;
            #pragma unroll
            for (int r = 0; r < 8; ++r)
                op[(size_t)r * L_] = acc[ct][lt][r];
        }
    }
}

// ---------------------------------------------------------------------------
extern "C" void kernel_launch(void* const* d_in, const int* in_sizes, int n_in,
                              void* d_out, int out_size, void* d_ws, size_t ws_size,
                              hipStream_t stream)
{
    const float* x      = (const float*)d_in[0];   // [16,256,8192]
    const float* style  = (const float*)d_in[1];   // [16,512]
    const float* weight = (const float*)d_in[2];   // [1,256,256,3]
    const float* mw     = (const float*)d_in[3];   // [256,512]
    const float* mb     = (const float*)d_in[4];   // [256]
    float* out = (float*)d_out;

    float*    s  = (float*)d_ws;                          // 16*256 f32 = 16 KB
    _Float16* wf = (_Float16*)((char*)d_ws + 16 * 1024);  // 16*256*3*256 f16 = 6.29 MB

    style_mod_kernel<<<B_, 256, 0, stream>>>(style, mw, mb, s);
    demod_kernel<<<(B_ * COUT_) / 8, 256, 0, stream>>>(weight, s, wf);

    dim3 grid(L_ / LBLK, B_);
    conv_wmma_kernel<<<grid, 256, 0, stream>>>(x, wf, out);
}
